// ChunkAligner_57226144252241
// MI455X (gfx1250) — compile-verified, matrix operand order checked by reading, not run on hardware
//
#include <hip/hip_runtime.h>

typedef __attribute__((ext_vector_type(16))) __bf16        v16bf;
typedef __attribute__((ext_vector_type(8)))  float         v8f;
typedef __attribute__((ext_vector_type(4)))  unsigned int  u32x4;
typedef __attribute__((ext_vector_type(4)))  unsigned short u16x4;
typedef __attribute__((ext_vector_type(4)))  float         f32x4;

union FragBF { v16bf v; u32x4 q[2]; };

__device__ __forceinline__ unsigned short f2bf(float f) {
    unsigned u = __float_as_uint(f);
    u += 0x7FFFu + ((u >> 16) & 1u);          // round-to-nearest-even
    return (unsigned short)(u >> 16);
}

// ---------------------------------------------------------------------------
// Kernel 1: pack W1 (fp32 [1024][512]) into bf16 WMMA-B fragment order:
//   w1pack[((ctile*32 + kstep)*32 + lane)*16 + i]
//   where n = ctile*16 + (lane&15), K = kstep*32 + (i/8)*16 + (lane>>4)*8 + (i&7)
// ---------------------------------------------------------------------------
__global__ __launch_bounds__(256) void pack_w1_kernel(
    const float* __restrict__ W1, unsigned short* __restrict__ w1pack)
{
    int g = blockIdx.x * 256 + threadIdx.x;   // 0..32767 fragments
    int lane  = g & 31;
    int kstep = (g >> 5) & 31;
    int ctile = g >> 10;                      // 0..31
    int n  = ctile * 16 + (lane & 15);
    int hs = (lane >> 4) & 1;
    unsigned short* dst = w1pack + (size_t)g * 16;
#pragma unroll
    for (int i = 0; i < 16; ++i) {
        int K = kstep * 32 + (i >> 3) * 16 + hs * 8 + (i & 7);
        dst[i] = f2bf(W1[(size_t)K * 512 + n]);
    }
}

// ---------------------------------------------------------------------------
// Kernel 2: fused  h = pf@W1+b1 -> LayerNorm -> GELU -> @W2+b2 -> scores[B*N]
// One block = 32 rows of M (M = B*N = 131072). 256 threads = 8 waves.
// LDS (64KB, overlaid): phase A = bf16 A-fragments, phase B = fp32 h-tile.
// ---------------------------------------------------------------------------
__global__ __launch_bounds__(256) void mlp_scores_kernel(
    const float* __restrict__ pf, const unsigned short* __restrict__ w1pack,
    const float* __restrict__ b1v, const float* __restrict__ gammav,
    const float* __restrict__ betav, const float* __restrict__ W2,
    const float* __restrict__ b2v, float* __restrict__ scores)
{
    __shared__ __align__(16) unsigned char smem[65536];
    unsigned short* Ap = (unsigned short*)smem;        // [2][32][32][16] bf16 frags
    float (*H)[512]    = (float (*)[512])smem;         // [32][512] fp32 (overlaid)

    const int tid = threadIdx.x;
    const size_t row0 = (size_t)blockIdx.x * 32;

    // ---- Phase 1: load 32x1024 fp32 tile, convert to bf16 in A-fragment order
    {
        int row = tid >> 3;                // 0..31
        int st  = row >> 4;                // M-subtile 0/1
        int m   = row & 15;
        int sub = tid & 7;
        const float* src = pf + (row0 + row) * 1024 + sub * 4;
#pragma unroll
        for (int j = 0; j < 32; ++j) {     // j == kstep; this thread covers kk=sub*4..+3
            f32x4 v = *(const f32x4*)(src + j * 32);
            int kk = sub * 4;
            int hs = (kk & 15) >> 3;
            int i0 = (kk >> 4) * 8 + (kk & 7);
            int lane = m + hs * 16;
            u16x4 p;
            p.x = f2bf(v.x); p.y = f2bf(v.y); p.z = f2bf(v.z); p.w = f2bf(v.w);
            *(u16x4*)&Ap[(((st * 32 + j) * 32) + lane) * 16 + i0] = p;
        }
    }
    __syncthreads();

    // ---- Phase 2: WMMA GEMM. wave w owns column tiles w*4..w*4+3, both M-subtiles.
    const int wave = tid >> 5;
    const int lane = tid & 31;
    const int mrow = lane & 15;
    const int hs   = (lane >> 4) & 1;

    v8f acc[2][4];
#pragma unroll
    for (int s = 0; s < 2; ++s)
#pragma unroll
        for (int c = 0; c < 4; ++c)
            acc[s][c] = (v8f){0.f,0.f,0.f,0.f,0.f,0.f,0.f,0.f};

    // B addressing: scalar base = w1pack + wave*128KB (uniform, via readfirstlane),
    // VGPR voffset = lane*32 (loop-invariant), (c,k) -> 24-bit instruction offset.
    const int waveS = __builtin_amdgcn_readfirstlane(wave);
    const char* __restrict__ wbase =
        (const char*)w1pack + (size_t)waveS * (4u * 32u * 32u * 32u); // 131072 B
    const int lane32 = lane * 32;

#pragma unroll 2
    for (int k = 0; k < 32; ++k) {
        FragBF a0, a1;
        const u32x4* a0p = (const u32x4*)&Ap[(((0 * 32 + k) * 32) + lane) * 16];
        const u32x4* a1p = (const u32x4*)&Ap[(((1 * 32 + k) * 32) + lane) * 16];
        a0.q[0] = a0p[0]; a0.q[1] = a0p[1];
        a1.q[0] = a1p[0]; a1.q[1] = a1p[1];
#pragma unroll
        for (int c = 0; c < 4; ++c) {
            const u32x4* bp =
                (const u32x4*)(wbase + (c * 32768 + k * 1024) + lane32);
            FragBF bm; bm.q[0] = bp[0]; bm.q[1] = bp[1];
            acc[0][c] = __builtin_amdgcn_wmma_f32_16x16x32_bf16(
                false, a0.v, false, bm.v, (short)0, acc[0][c], false, false);
            acc[1][c] = __builtin_amdgcn_wmma_f32_16x16x32_bf16(
                false, a1.v, false, bm.v, (short)0, acc[1][c], false, false);
        }
    }
    __syncthreads();   // A-fragments dead; safe to overlay H

    // ---- Phase 3: spill h (+b1) to LDS. C/D layout: lane -> N=lane&15, M=r+hs*8.
#pragma unroll
    for (int s = 0; s < 2; ++s) {
#pragma unroll
        for (int c = 0; c < 4; ++c) {
            int n = (wave * 4 + c) * 16 + mrow;
            float bb = b1v[n];
#pragma unroll
            for (int r = 0; r < 8; ++r)
                H[s * 16 + hs * 8 + r][n] = acc[s][c][r] + bb;
        }
    }
    __syncthreads();

    // ---- Phase 4: LayerNorm + exact GELU + dot(W2). 8 threads per row.
    {
        int row = tid >> 3;                // 0..31
        int sub = tid & 7;
        float sum = 0.f, ssq = 0.f;
#pragma unroll 8
        for (int j = 0; j < 64; ++j) {
            float v = H[row][sub + j * 8];
            sum += v; ssq += v * v;
        }
#pragma unroll
        for (int msk = 4; msk >= 1; msk >>= 1) {
            sum += __shfl_xor(sum, msk, 32);
            ssq += __shfl_xor(ssq, msk, 32);
        }
        float mean = sum * (1.0f / 512.0f);
        float var  = ssq * (1.0f / 512.0f) - mean * mean;
        float inv  = rsqrtf(var + 1e-5f);
        float part = 0.f;
#pragma unroll 8
        for (int j = 0; j < 64; ++j) {
            int n = sub + j * 8;
            float v = (H[row][n] - mean) * inv * gammav[n] + betav[n];
            float g = 0.5f * v * (1.0f + erff(v * 0.70710678118654752f));
            part += g * W2[n];
        }
#pragma unroll
        for (int msk = 4; msk >= 1; msk >>= 1)
            part += __shfl_xor(part, msk, 32);
        if (sub == 0)
            scores[row0 + row] = part + b2v[0];
    }
}

// ---------------------------------------------------------------------------
// Kernel 3: per-b softmaxes + weighted feature sum. One block per b.
// ---------------------------------------------------------------------------
__global__ __launch_bounds__(256) void combine_kernel(
    const float* __restrict__ pf, const float* __restrict__ scores,
    const int* __restrict__ cposp, const int* __restrict__ tlenp,
    float* __restrict__ out)
{
    __shared__ float red[256];
    __shared__ float wbuf[256];
    const int b = blockIdx.x;
    const int t = threadIdx.x;

    int cp = cposp[0], tl = tlenp[0];
    int ce = cp + 32; if (ce > tl) ce = tl;
    float prog = ((float)cp + 0.5f * (float)(ce - cp)) / (float)tl;

    float rowp = (float)(t >> 4) * (1.0f / 15.0f);
    float colp = (float)(t & 15) * (1.0f / 15.0f);
    float sraw = expf(-fabsf(rowp * 0.7f + colp * 0.3f - prog) * 3.0f);

    auto bmax = [&](float v) -> float {
        red[t] = v; __syncthreads();
        for (int s = 128; s > 0; s >>= 1) {
            if (t < s) red[t] = fmaxf(red[t], red[t + s]);
            __syncthreads();
        }
        float r = red[0]; __syncthreads(); return r;
    };
    auto bsum = [&](float v) -> float {
        red[t] = v; __syncthreads();
        for (int s = 128; s > 0; s >>= 1) {
            if (t < s) red[t] = red[t] + red[t + s];
            __syncthreads();
        }
        float r = red[0]; __syncthreads(); return r;
    };

    // spatial = softmax(exp(-|bias-prog|*3))      (exp-then-softmax, as in ref)
    float e1 = expf(sraw - bmax(sraw));
    float spatial = e1 / bsum(e1);

    // learned = softmax(scores[b, :])
    float sc = scores[(size_t)b * 256 + t];
    float e2 = expf(sc - bmax(sc));
    float learned = e2 / bsum(e2);

    // combined = softmax(0.7*spatial + 0.3*learned)
    float cb = 0.7f * spatial + 0.3f * learned;
    float e3 = expf(cb - bmax(cb));
    float w  = e3 / bsum(e3);
    wbuf[t] = w;
    __syncthreads();

    // out[b, :] = sum_n pf[b, n, :] * w[n]   (each thread owns one float4 column)
    const f32x4* p4 = (const f32x4*)(pf + (size_t)b * (256 * 1024));
    f32x4 acc = (f32x4){0.f, 0.f, 0.f, 0.f};
#pragma unroll 4
    for (int n = 0; n < 256; ++n) {
        float wn = wbuf[n];
        f32x4 v = p4[(size_t)n * 256 + t];
        acc += wn * v;
    }
    ((f32x4*)out)[(size_t)b * 256 + t] = acc;
}

// ---------------------------------------------------------------------------
extern "C" void kernel_launch(void* const* d_in, const int* in_sizes, int n_in,
                              void* d_out, int out_size, void* d_ws, size_t ws_size,
                              hipStream_t stream)
{
    const float* pf    = (const float*)d_in[0];
    const float* W1    = (const float*)d_in[1];
    const float* b1v   = (const float*)d_in[2];
    const float* gamma = (const float*)d_in[3];
    const float* beta  = (const float*)d_in[4];
    const float* W2    = (const float*)d_in[5];
    const float* b2v   = (const float*)d_in[6];
    const int*   cpos  = (const int*)d_in[7];
    const int*   tlen  = (const int*)d_in[8];
    float* out = (float*)d_out;

    unsigned short* w1pack = (unsigned short*)d_ws;                 // 1 MB
    float* scores = (float*)((char*)d_ws + (1u << 20));             // B*N fp32

    const int M     = in_sizes[0] / 1024;   // B*N
    const int B     = M / 256;
    const int tiles = M / 32;

    pack_w1_kernel<<<128, 256, 0, stream>>>(W1, w1pack);
    mlp_scores_kernel<<<tiles, 256, 0, stream>>>(pf, w1pack, b1v, gamma, beta,
                                                 W2, b2v, scores);
    combine_kernel<<<B, 256, 0, stream>>>(pf, scores, cpos, tlen, out);
}
